// GNNConnectivityEncoder_80977313399245
// MI455X (gfx1250) — compile-verified, matrix operand order checked by reading, not run on hardware
//
#include <hip/hip_runtime.h>

// ---------------- problem constants ----------------
#define BB     64
#define NN     256
#define CT     3
#define F_IN   (NN * CT)     // 768
#define HID    256
#define HEADS  4
#define CH     (HID / HEADS) // 64
#define EDGES  8192
#define NEG_SLOPE 0.2f

// ---------------- WMMA types ----------------
typedef __attribute__((ext_vector_type(16))) __bf16   v16bf;
typedef __attribute__((ext_vector_type(8)))  float    v8f;
typedef __attribute__((ext_vector_type(4))) unsigned int u32x4;
typedef __attribute__((ext_vector_type(4)))  float    f32x4;

struct Frag256 { u32x4 lo, hi; };   // 32 bytes == v16bf

__device__ __forceinline__ unsigned short f2bf(float f) {
    unsigned int u = __float_as_uint(f);
    unsigned int r = (u + 0x7fffu + ((u >> 16) & 1u)) >> 16;  // RNE
    return (unsigned short)r;
}
__device__ __forceinline__ unsigned int pk2(float a, float b) {
    return (unsigned int)f2bf(a) | ((unsigned int)f2bf(b) << 16);
}
__device__ __forceinline__ float gelu_exact(float x) {
    return 0.5f * x * (1.0f + erff(x * 0.70710678118654752f));
}

// Per-lane A/B fragment for V_WMMA_F32_16X16X32_BF16.
// 16-bit striping (ISA 7.12.2): lanes 0-15 hold K {kb..kb+7, kb+16..kb+23}
// with kb = k0 (lanes<16) or k0+8 (lanes>=16); two contiguous 16B runs.
__device__ __forceinline__ v16bf frag_bf16(const unsigned short* p) {
    Frag256 f;
    f.lo = *(const u32x4*)(p);
    f.hi = *(const u32x4*)(p + 16);
    return __builtin_bit_cast(v16bf, f);
}

#define WMMA_BF16(d, a, b) \
    d = __builtin_amdgcn_wmma_f32_16x16x32_bf16(false, a, false, b, (short)0, d, false, false)

// ---------------- elementwise f32 -> bf16 (8 elems / thread) ----------------
__global__ void cvt_bf16_kernel(const float* __restrict__ in,
                                unsigned short* __restrict__ out, int n8) {
    int t = blockIdx.x * blockDim.x + threadIdx.x;
    if (t >= n8) return;
    const f32x4* p = (const f32x4*)(in + (size_t)t * 8);
    f32x4 a = p[0], b = p[1];
    u32x4 o = { pk2(a.x, a.y), pk2(a.z, a.w), pk2(b.x, b.y), pk2(b.z, b.w) };
    *(u32x4*)(out + (size_t)t * 8) = o;
}

// ---------------- weight convert: W[K][N] f32 -> Wt[N][K] bf16 ----------------
__global__ void wt_convert_kernel(const float* __restrict__ W,
                                  unsigned short* __restrict__ Wt,
                                  int K, int Nc) {
    int idx = blockIdx.x * blockDim.x + threadIdx.x;
    if (idx >= K * Nc) return;
    int n = idx / K, k = idx - n * K;
    Wt[idx] = f2bf(W[(size_t)k * Nc + n]);
}

// ---------------- deterministic CSR build (1 block, 256 threads) ----------------
__global__ void build_csr_kernel(const int* __restrict__ ei,   // [2][E] (src,dst)
                                 int* __restrict__ rowptr,     // [N+1]
                                 int* __restrict__ colsrc) {   // [E]
    __shared__ int cnt[NN];
    __shared__ int pfx[NN + 1];
    int t = threadIdx.x;                // node id, 256 threads
    const int* src = ei;
    const int* dst = ei + EDGES;
    int c = 0;
    for (int e = 0; e < EDGES; ++e) c += (dst[e] == t);
    cnt[t] = c;
    __syncthreads();
    if (t == 0) {
        int s = 0; pfx[0] = 0;
        for (int n = 0; n < NN; ++n) { s += cnt[n]; pfx[n + 1] = s; }
    }
    __syncthreads();
    rowptr[t] = pfx[t];
    if (t == 0) rowptr[NN] = pfx[NN];
    int o = pfx[t];
    for (int e = 0; e < EDGES; ++e)     // stable order: edge index ascending
        if (dst[e] == t) colsrc[o++] = src[e];
}

// ---------------- WMMA GEMM: out = epilogue(A[M,K] x Wt^T + bias) ----------------
// A is [M][K] bf16, Wt is [Nc][K] bf16. 4 waves/block arranged 2x2,
// 128x128 block tile, 64x64 per wave = 4x4 WMMA accumulators.
// Inner K step: 16x global_load_b128 feeding 16x v_wmma (32 FLOP/B from L2).
__global__ __launch_bounds__(128) void gemm_wmma_kernel(
    const unsigned short* __restrict__ A,
    const unsigned short* __restrict__ Wt,
    const float* __restrict__ bias,
    const float* __restrict__ bng, const float* __restrict__ bnb,
    const float* __restrict__ bnm, const float* __restrict__ bnv,
    int M, int K, int Nc, int do_gelu,
    float* __restrict__ out_f32, unsigned short* __restrict__ out_bf16)
{
    const int lane  = threadIdx.x & 31;
    const int wave  = threadIdx.x >> 5;
    const int mw    = (wave >> 1) * 64;
    const int nw    = (wave & 1) * 64;
    const int blkM  = blockIdx.y * 128;
    const int blkN  = blockIdx.x * 128;
    const int half  = lane >> 4;        // 0|1
    const int l15   = lane & 15;
    const int koff  = half * 8;

    // named scalar base pointers (keeps address-space inference -> global_load_b128)
    const unsigned short* pa0 = A  + (size_t)(blkM + mw      + l15) * K + koff;
    const unsigned short* pa1 = A  + (size_t)(blkM + mw + 16 + l15) * K + koff;
    const unsigned short* pa2 = A  + (size_t)(blkM + mw + 32 + l15) * K + koff;
    const unsigned short* pa3 = A  + (size_t)(blkM + mw + 48 + l15) * K + koff;
    const unsigned short* pb0 = Wt + (size_t)(blkN + nw      + l15) * K + koff;
    const unsigned short* pb1 = Wt + (size_t)(blkN + nw + 16 + l15) * K + koff;
    const unsigned short* pb2 = Wt + (size_t)(blkN + nw + 32 + l15) * K + koff;
    const unsigned short* pb3 = Wt + (size_t)(blkN + nw + 48 + l15) * K + koff;

    v8f acc[4][4] = {};
    for (int k0 = 0; k0 < K; k0 += 32) {
        v16bf b0 = frag_bf16(pb0); pb0 += 32;
        v16bf b1 = frag_bf16(pb1); pb1 += 32;
        v16bf b2 = frag_bf16(pb2); pb2 += 32;
        v16bf b3 = frag_bf16(pb3); pb3 += 32;
        v16bf a0 = frag_bf16(pa0); pa0 += 32;
        WMMA_BF16(acc[0][0], a0, b0);
        WMMA_BF16(acc[0][1], a0, b1);
        WMMA_BF16(acc[0][2], a0, b2);
        WMMA_BF16(acc[0][3], a0, b3);
        v16bf a1 = frag_bf16(pa1); pa1 += 32;
        WMMA_BF16(acc[1][0], a1, b0);
        WMMA_BF16(acc[1][1], a1, b1);
        WMMA_BF16(acc[1][2], a1, b2);
        WMMA_BF16(acc[1][3], a1, b3);
        v16bf a2 = frag_bf16(pa2); pa2 += 32;
        WMMA_BF16(acc[2][0], a2, b0);
        WMMA_BF16(acc[2][1], a2, b1);
        WMMA_BF16(acc[2][2], a2, b2);
        WMMA_BF16(acc[2][3], a2, b3);
        v16bf a3 = frag_bf16(pa3); pa3 += 32;
        WMMA_BF16(acc[3][0], a3, b0);
        WMMA_BF16(acc[3][1], a3, b1);
        WMMA_BF16(acc[3][2], a3, b2);
        WMMA_BF16(acc[3][3], a3, b3);
    }

    // epilogue: element v of acc tile sits at (M = v + half*8, N = l15).
    // M is a multiple of 16 in all launches -> per-tile row guard only.
    const bool hasBN  = (bng != nullptr);
    const bool hasF32 = (out_f32 != nullptr);
    const bool hasBF  = (out_bf16 != nullptr);
#pragma unroll
    for (int j = 0; j < 4; ++j) {
        const int cN = blkN + nw + j * 16 + l15;
        const float bv = bias[cN];
        float sc = 1.0f, mm = 0.0f, bb = 0.0f;
        if (hasBN) {
            sc = bng[cN] * rsqrtf(bnv[cN] + 1e-5f);
            mm = bnm[cN];
            bb = bnb[cN];
        }
#pragma unroll
        for (int i = 0; i < 4; ++i) {
            const int rbase = blkM + mw + i * 16;
            if (rbase < M) {
#pragma unroll
                for (int v = 0; v < 8; ++v) {
                    const int r = rbase + half * 8 + v;
                    float val = acc[i][j][v] + bv;
                    if (hasBN) val = (val - mm) * sc + bb;
                    if (do_gelu) val = gelu_exact(val);
                    if (hasF32) out_f32[(size_t)r * Nc + cN] = val;
                    if (hasBF)  out_bf16[(size_t)r * Nc + cN] = f2bf(val);
                }
            }
        }
    }
}

// ---------------- GATv2 edge softmax + aggregation ----------------
// Block per (sample b, node i); wave h handles head h (2 channels per lane).
__global__ __launch_bounds__(128) void gat_edge_kernel(
    const float* __restrict__ xl,      // [B*N][HID] lin_l(h) (source features)
    const float* __restrict__ xr,      // [B*N][HID] lin_r(h) (target features)
    const float* __restrict__ att,     // [HEADS][CH]
    const float* __restrict__ bo,      // [HID]
    const int* __restrict__ rowptr, const int* __restrict__ colsrc,
    float* __restrict__ out_f32, unsigned short* __restrict__ out_bf16)
{
    const int b    = blockIdx.x >> 8;
    const int i    = blockIdx.x & 255;
    const int lane = threadIdx.x & 31;
    const int h    = threadIdx.x >> 5;
    const int c0   = lane * 2;

    const size_t nodeI = (size_t)(b * NN + i) * HID + h * CH;
    const float xri0 = xr[nodeI + c0];
    const float xri1 = xr[nodeI + c0 + 1];
    const float a0 = att[h * CH + c0];
    const float a1 = att[h * CH + c0 + 1];
    const int e0 = rowptr[i], e1 = rowptr[i + 1];

    // pass 1: neighborhood max of attention logits
    float mx = -3.4e38f;
    for (int e = e0; e < e1; ++e) {
        const int s = colsrc[e];
        const size_t nodeS = (size_t)(b * NN + s) * HID + h * CH;
        float v0 = xri0 + xl[nodeS + c0];
        float v1 = xri1 + xl[nodeS + c0 + 1];
        v0 = v0 > 0.0f ? v0 : NEG_SLOPE * v0;
        v1 = v1 > 0.0f ? v1 : NEG_SLOPE * v1;
        float p = a0 * v0 + a1 * v1;
#pragma unroll
        for (int m = 1; m < 32; m <<= 1) p += __shfl_xor(p, m, 32);
        mx = fmaxf(mx, p);
    }

    // pass 2: exp / sum / weighted aggregation of raw x_j
    float den = 0.0f, s0 = 0.0f, s1 = 0.0f;
    for (int e = e0; e < e1; ++e) {
        const int s = colsrc[e];
        const size_t nodeS = (size_t)(b * NN + s) * HID + h * CH;
        const float x0 = xl[nodeS + c0];
        const float x1 = xl[nodeS + c0 + 1];
        float v0 = xri0 + x0; v0 = v0 > 0.0f ? v0 : NEG_SLOPE * v0;
        float v1 = xri1 + x1; v1 = v1 > 0.0f ? v1 : NEG_SLOPE * v1;
        float p = a0 * v0 + a1 * v1;
#pragma unroll
        for (int m = 1; m < 32; m <<= 1) p += __shfl_xor(p, m, 32);
        const float w = expf(p - mx);
        den += w; s0 += w * x0; s1 += w * x1;
    }
    const float inv = 1.0f / (den + 1e-16f);
    float o0 = gelu_exact(s0 * inv + bo[h * CH + c0]);
    float o1 = gelu_exact(s1 * inv + bo[h * CH + c0 + 1]);
    if (out_f32)  { out_f32[nodeI + c0] = o0;  out_f32[nodeI + c0 + 1] = o1; }
    if (out_bf16) { out_bf16[nodeI + c0] = f2bf(o0); out_bf16[nodeI + c0 + 1] = f2bf(o1); }
}

// ---------------- mean pool over nodes -> bf16 (pads rows BB..127 with 0) ----------
__global__ void mean_pool_kernel(const float* __restrict__ out1,
                                 unsigned short* __restrict__ pooled_bf16) {
    const int b = blockIdx.x, t = threadIdx.x;  // grid 128, 256 threads = HID channels
    float s = 0.0f;
    if (b < BB)
        for (int n = 0; n < NN; ++n) s += out1[(size_t)(b * NN + n) * HID + t];
    pooled_bf16[b * HID + t] = f2bf(b < BB ? s * (1.0f / (float)NN) : 0.0f);
}

// ---------------- launch ----------------
extern "C" void kernel_launch(void* const* d_in, const int* in_sizes, int n_in,
                              void* d_out, int out_size, void* d_ws, size_t ws_size,
                              hipStream_t stream) {
    (void)in_sizes; (void)n_in; (void)out_size; (void)ws_size;
    const float* x    = (const float*)d_in[0];
    const int*   ei   = (const int*)  d_in[1];
    const float* W1   = (const float*)d_in[2];
    const float* b1   = (const float*)d_in[3];
    const float* g1p  = (const float*)d_in[4];
    const float* be1  = (const float*)d_in[5];
    const float* m1   = (const float*)d_in[6];
    const float* v1   = (const float*)d_in[7];
    const float* Wl0  = (const float*)d_in[8];
    const float* bl0  = (const float*)d_in[9];
    const float* Wr0  = (const float*)d_in[10];
    const float* br0  = (const float*)d_in[11];
    const float* att0 = (const float*)d_in[12];
    const float* bo0  = (const float*)d_in[13];
    const float* Wl1  = (const float*)d_in[14];
    const float* bl1  = (const float*)d_in[15];
    const float* Wr1  = (const float*)d_in[16];
    const float* br1  = (const float*)d_in[17];
    const float* att1 = (const float*)d_in[18];
    const float* bo1  = (const float*)d_in[19];
    const float* W2   = (const float*)d_in[20];
    const float* b2   = (const float*)d_in[21];
    const float* g2p  = (const float*)d_in[22];
    const float* be2  = (const float*)d_in[23];
    const float* m2   = (const float*)d_in[24];
    const float* v2   = (const float*)d_in[25];
    float* outp = (float*)d_out;

    char* ws = (char*)d_ws;
    size_t off = 0;
    auto alloc = [&](size_t bytes) -> char* {
        char* p = ws + off;
        off += (bytes + 255) & ~(size_t)255;
        return p;
    };
    const int M = BB * NN;  // 16384
    unsigned short* W1t  = (unsigned short*)alloc((size_t)F_IN * HID * 2);
    unsigned short* Wl0t = (unsigned short*)alloc((size_t)HID * HID * 2);
    unsigned short* Wr0t = (unsigned short*)alloc((size_t)HID * HID * 2);
    unsigned short* Wl1t = (unsigned short*)alloc((size_t)HID * HID * 2);
    unsigned short* Wr1t = (unsigned short*)alloc((size_t)HID * HID * 2);
    unsigned short* W2t  = (unsigned short*)alloc((size_t)HID * HID * 2);
    int*            rowptr = (int*)alloc((NN + 1) * 4);
    int*            colsrc = (int*)alloc(EDGES * 4);
    unsigned short* xbf    = (unsigned short*)alloc((size_t)M * F_IN * 2);
    unsigned short* hbf    = (unsigned short*)alloc((size_t)M * HID * 2);
    float*          xlbuf  = (float*)alloc((size_t)M * HID * 4);
    float*          xrbuf  = (float*)alloc((size_t)M * HID * 4);
    unsigned short* act0   = (unsigned short*)alloc((size_t)M * HID * 2);
    float*          out1   = (float*)alloc((size_t)M * HID * 4);
    unsigned short* pooled = (unsigned short*)alloc((size_t)128 * HID * 2); // padded to 128 rows

    // 0) x -> bf16 (one memory-bound pass; keeps GEMM inner loops pure WMMA)
    {
        const int n8 = (M * F_IN) / 8;  // 1,572,864
        cvt_bf16_kernel<<<(n8 + 255) / 256, 256, 0, stream>>>(x, xbf, n8);
    }

    // 1) weights -> bf16 [N][K]
    wt_convert_kernel<<<(F_IN * HID + 255) / 256, 256, 0, stream>>>(W1, W1t, F_IN, HID);
    wt_convert_kernel<<<(HID * HID + 255) / 256, 256, 0, stream>>>(Wl0, Wl0t, HID, HID);
    wt_convert_kernel<<<(HID * HID + 255) / 256, 256, 0, stream>>>(Wr0, Wr0t, HID, HID);
    wt_convert_kernel<<<(HID * HID + 255) / 256, 256, 0, stream>>>(Wl1, Wl1t, HID, HID);
    wt_convert_kernel<<<(HID * HID + 255) / 256, 256, 0, stream>>>(Wr1, Wr1t, HID, HID);
    wt_convert_kernel<<<(HID * HID + 255) / 256, 256, 0, stream>>>(W2, W2t, HID, HID);

    // 2) deterministic CSR over dst
    build_csr_kernel<<<1, 256, 0, stream>>>(ei, rowptr, colsrc);

    const dim3 gemmGridBig(HID / 128, M / 128);   // (2, 128)
    const dim3 gemmGridOut(HID / 128, 1);         // (2, 1), M=64 guarded

    // 3) node_proj: h = gelu(BN(x @ W1 + b1)) -> bf16
    gemm_wmma_kernel<<<gemmGridBig, 128, 0, stream>>>(
        xbf, W1t, b1, g1p, be1, m1, v1, M, F_IN, HID, 1, nullptr, hbf);

    // 4) layer 0: xl = h@Wl0+bl0, xr = h@Wr0+br0 (f32)
    gemm_wmma_kernel<<<gemmGridBig, 128, 0, stream>>>(
        hbf, Wl0t, bl0, nullptr, nullptr, nullptr, nullptr, M, HID, HID, 0, xlbuf, nullptr);
    gemm_wmma_kernel<<<gemmGridBig, 128, 0, stream>>>(
        hbf, Wr0t, br0, nullptr, nullptr, nullptr, nullptr, M, HID, HID, 0, xrbuf, nullptr);
    gat_edge_kernel<<<M, 128, 0, stream>>>(
        xlbuf, xrbuf, att0, bo0, rowptr, colsrc, nullptr, act0);

    // 5) layer 1
    gemm_wmma_kernel<<<gemmGridBig, 128, 0, stream>>>(
        act0, Wl1t, bl1, nullptr, nullptr, nullptr, nullptr, M, HID, HID, 0, xlbuf, nullptr);
    gemm_wmma_kernel<<<gemmGridBig, 128, 0, stream>>>(
        act0, Wr1t, br1, nullptr, nullptr, nullptr, nullptr, M, HID, HID, 0, xrbuf, nullptr);
    gat_edge_kernel<<<M, 128, 0, stream>>>(
        xlbuf, xrbuf, att1, bo1, rowptr, colsrc, out1, nullptr);

    // 6) mean over nodes -> bf16 pooled (rows 64..127 zero-padded)
    mean_pool_kernel<<<128, 256, 0, stream>>>(out1, pooled);

    // 7) output_proj: gelu(BN(pooled @ W2 + b2)) -> d_out (f32)
    gemm_wmma_kernel<<<gemmGridOut, 128, 0, stream>>>(
        pooled, W2t, b2, g2p, be2, m2, v2, BB, HID, HID, 1, outp, nullptr);
}
